// dense_mincut_pool_31121333027153
// MI455X (gfx1250) — matrix operand order repaired and checked by reading, not compiled, not run on hardware
//
#include <hip/hip_runtime.h>

typedef unsigned short u16;
typedef unsigned int   u32;
typedef __attribute__((ext_vector_type(16))) __bf16 v16bf;
typedef __attribute__((ext_vector_type(8)))  float  v8f;

static constexpr int BB = 8;      // batches
static constexpr int NN = 2000;   // nodes
static constexpr int NP = 2048;   // padded nodes (zero filled)
static constexpr int KC = 64;     // clusters
static constexpr int FF = 128;    // features
static constexpr int KEND = 2016; // 63 chunks of 32 (chunk 63 is all padding)

#if __has_builtin(__builtin_amdgcn_global_load_async_to_lds_b128) && \
    __has_builtin(__builtin_amdgcn_s_wait_asynccnt)
#define HAVE_ASYNC 1
typedef int v4i_t __attribute__((vector_size(16)));
typedef __attribute__((address_space(1))) v4i_t gv4i;   // global int4*
typedef __attribute__((address_space(3))) v4i_t lv4i;   // LDS int4*
#else
#define HAVE_ASYNC 0
#endif

union Frag { u32 u[8]; v16bf v; };

__device__ __forceinline__ u16 f2bf(float x) {
    u32 u = __float_as_uint(x);
    return (u16)((u + 0x7FFFu + ((u >> 16) & 1u)) >> 16);
}
__device__ __forceinline__ float bf2f(u16 h) { return __uint_as_float(((u32)h) << 16); }
__device__ __forceinline__ void split2(float x, float y, u32& uh, u32& ul) {
    u16 hx = f2bf(x), hy = f2bf(y);
    uh = (u32)hx | ((u32)hy << 16);
    ul = (u32)f2bf(x - bf2f(hx)) | ((u32)f2bf(y - bf2f(hy)) << 16);
}

// D = A*B + C with split bf16 (drop al*bl term, ~2^-16 relative)
__device__ __forceinline__ v8f wmma3(const Frag& ah, const Frag& al,
                                     const Frag& bh, const Frag& bl, v8f c) {
    c = __builtin_amdgcn_wmma_f32_16x16x32_bf16(false, ah.v, false, bh.v, (short)0, c, false, false);
    c = __builtin_amdgcn_wmma_f32_16x16x32_bf16(false, ah.v, false, bl.v, (short)0, c, false, false);
    c = __builtin_amdgcn_wmma_f32_16x16x32_bf16(false, al.v, false, bh.v, (short)0, c, false, false);
    return c;
}

// A-matrix 16x32 bf16 fragment: lane&15 = M row; K = (lane&16) + 2i + h
__device__ __forceinline__ void fragA_load(const u16* p, int lane, Frag& f) {
    int koff = lane & 16;
#pragma unroll
    for (int i = 0; i < 8; ++i) {
        int k = koff + (i << 1);
        f.u[i] = *(const u32*)(p + k);
    }
}
// B-matrix 32x16 bf16 fragment: lane&15 = N col; K = (lane&16) + 2i + h
__device__ __forceinline__ void fragB_load(const u16* p, int lane, Frag& f) {
    int koff = lane & 16;
#pragma unroll
    for (int i = 0; i < 8; ++i)
        f.u[i] = *(const u32*)(p + koff + (i << 1));
}

__device__ __forceinline__ float block_reduce(float v, float* red, int t) {
    red[t] = v;
    __syncthreads();
    for (int s = 128; s > 0; s >>= 1) {
        if (t < s) red[t] += red[t + s];
        __syncthreads();
    }
    float r = red[0];
    __syncthreads();
    return r;
}

// ---------------------------------------------------------------- init
__global__ void k_init(float* d_out) {
    if (threadIdx.x < 2) d_out[98304 + threadIdx.x] = 0.0f;
}

// ---------------------------------------------------------------- softmax + sum(s^2)
__global__ __launch_bounds__(256) void k_softmax(const float* __restrict__ s,
                                                 const unsigned char* __restrict__ mask,
                                                 float* __restrict__ s_soft,
                                                 float* __restrict__ sden) {
    int t = threadIdx.x, lane = t & 31, w = t >> 5;
    int row = blockIdx.x * 8 + w;            // [0, 16000)
    int b = row / NN, n = row - b * NN;
    const float* sp = s + (size_t)row * KC;
    float v0 = sp[lane], v1 = sp[lane + 32];
    float m = fmaxf(v0, v1);
#pragma unroll
    for (int off = 16; off > 0; off >>= 1) m = fmaxf(m, __shfl_xor(m, off, 32));
    float e0 = expf(v0 - m), e1 = expf(v1 - m);
    float sum = e0 + e1;
#pragma unroll
    for (int off = 16; off > 0; off >>= 1) sum += __shfl_xor(sum, off, 32);
    float mval = mask[row] ? 1.0f : 0.0f;
    float inv = mval / sum;
    float p0 = e0 * inv, p1 = e1 * inv;
    s_soft[(size_t)row * KC + lane]      = p0;
    s_soft[(size_t)row * KC + lane + 32] = p1;
    float q = p0 * p0 + p1 * p1;
#pragma unroll
    for (int off = 16; off > 0; off >>= 1) q += __shfl_xor(q, off, 32);
    if (lane == 0) sden[(size_t)b * NP + n] = q;
}

// ---------------------------------------------------------------- transpose s -> sT (bf16 hi/lo, padded)
__global__ __launch_bounds__(256) void k_transpose_s(const float* __restrict__ s_soft,
                                                     u16* __restrict__ sT_h,
                                                     u16* __restrict__ sT_l) {
    __shared__ float tile[32][33];
    int b = blockIdx.z, bx = blockIdx.x, by = blockIdx.y, t = threadIdx.x;
#pragma unroll
    for (int j = 0; j < 4; ++j) {
        int idx = t + (j << 8);
        int nl = idx >> 5, kl = idx & 31;
        int gn = (bx << 5) + nl, gk = (by << 5) + kl;
        tile[nl][kl] = (gn < NN) ? s_soft[((size_t)(b * NN + gn)) * KC + gk] : 0.0f;
    }
    __syncthreads();
#pragma unroll
    for (int j = 0; j < 4; ++j) {
        int idx = t + (j << 8);
        int kl = idx >> 5, nl = idx & 31;
        int gk = (by << 5) + kl, gn = (bx << 5) + nl;
        float v = tile[nl][kl];
        size_t o = (size_t)(b * KC + gk) * NP + gn;
        u16 h = f2bf(v);
        sT_h[o] = h;
        sT_l[o] = f2bf(v - bf2f(h));
    }
}

// ---------------------------------------------------------------- transpose x*mask -> xT (bf16 hi/lo, padded)
__global__ __launch_bounds__(256) void k_transpose_x(const float* __restrict__ x,
                                                     const unsigned char* __restrict__ mask,
                                                     u16* __restrict__ xT_h,
                                                     u16* __restrict__ xT_l) {
    __shared__ float tile[32][33];
    int b = blockIdx.z, bx = blockIdx.x, by = blockIdx.y, t = threadIdx.x;
#pragma unroll
    for (int j = 0; j < 4; ++j) {
        int idx = t + (j << 8);
        int nl = idx >> 5, kl = idx & 31;
        int gn = (bx << 5) + nl, gf = (by << 5) + kl;
        float v = 0.0f;
        if (gn < NN && mask[b * NN + gn])
            v = x[((size_t)(b * NN + gn)) * FF + gf];
        tile[nl][kl] = v;
    }
    __syncthreads();
#pragma unroll
    for (int j = 0; j < 4; ++j) {
        int idx = t + (j << 8);
        int kl = idx >> 5, nl = idx & 31;
        int gf = (by << 5) + kl, gn = (bx << 5) + nl;
        float v = tile[nl][kl];
        size_t o = (size_t)(b * FF + gf) * NP + gn;
        u16 h = f2bf(v);
        xT_h[o] = h;
        xT_l[o] = f2bf(v - bf2f(h));
    }
}

// ---------------------------------------------------------------- as = adj @ s  (writes asT bf16 hi/lo) + d_flat
// grid (32 strips of 64 rows, 8 batches); 8 waves: wave = (colhalf<<2)|rowtile, 16x32 output per wave.
// Double-buffered LDS staging; async-to-LDS (ASYNCcnt) when available, checked sync loads at boundaries.

// stage one 32-wide chunk into buffer; returns # async ops issued by this thread
__device__ __forceinline__ int stage_chunk(int c, int t, int strip, int b, bool strip_ok,
                                           const float* __restrict__ adj, size_t adjb,
                                           const u16* __restrict__ sT_h,
                                           const u16* __restrict__ sT_l,
                                           float* AsmB, u16* BhB, u16* BlB) {
    const int kk = c << 5;
    int pend = 0;
#if HAVE_ASYNC
    // B (s^T chunk): source is zero-padded to NP -> always async-safe
#pragma unroll
    for (int q = 0; q < 2; ++q) {
        int idx = t + (q << 8);            // [0,512)
        int arr = idx >> 8, ii = idx & 255;
        int row = ii >> 2, c8 = (ii & 3) << 3;
        const u16* src = (arr ? sT_l : sT_h) + (size_t)(b * KC + row) * NP + kk + c8;
        u16* dst = (arr ? BlB : BhB) + row * 40 + c8;
        __builtin_amdgcn_global_load_async_to_lds_b128((gv4i*)src, (lv4i*)dst, 0, 0);
    }
    pend += 2;
    if (strip_ok && c < 62) {              // fully interior adj tile
#pragma unroll
        for (int q = 0; q < 2; ++q) {
            int idx = t + (q << 8);        // [0,512)
            int r = idx >> 3, c4 = (idx & 7) << 2;
            const float* gp = adj + adjb + (size_t)((strip << 6) + r) * NN + kk + c4;
            float* lp = AsmB + r * 36 + c4;
            __builtin_amdgcn_global_load_async_to_lds_b128((gv4i*)gp, (lv4i*)lp, 0, 0);
        }
        pend += 2;
    } else
#endif
    {
        // checked synchronous adj staging (boundaries / no-async fallback)
#pragma unroll
        for (int q = 0; q < 2; ++q) {
            int idx = t + (q << 8);
            int r = idx >> 3, c4 = (idx & 7) << 2;
            int grow = (strip << 6) + r;
            int gcol = kk + c4;
            float4 v = {0.f, 0.f, 0.f, 0.f};
            if (grow < NN) {
                const float* src = adj + adjb + (size_t)grow * NN;
                if (gcol + 3 < NN) {
                    v = *(const float4*)(src + gcol);
                } else {
                    if (gcol     < NN) v.x = src[gcol];
                    if (gcol + 1 < NN) v.y = src[gcol + 1];
                    if (gcol + 2 < NN) v.z = src[gcol + 2];
                }
            }
            *(float4*)(AsmB + r * 36 + c4) = v;
        }
    }
#if !HAVE_ASYNC
    // synchronous B staging
#pragma unroll
    for (int j = 0; j < 8; ++j) {
        int idx = t + (j << 8);
        int arr = idx >> 10, ii = idx & 1023;
        int rowc = ii >> 4, c2 = (ii & 15) << 1;
        const u16* src = arr ? sT_l : sT_h;
        u32 val = *(const u32*)(src + (size_t)(b * KC + rowc) * NP + kk + c2);
        u16* dst = arr ? BlB : BhB;
        *(u32*)(dst + rowc * 40 + c2) = val;
    }
#endif
    return pend;
}

__global__ __launch_bounds__(256) void k_adj_s(const float* __restrict__ adj,
                                               const u16* __restrict__ sT_h,
                                               const u16* __restrict__ sT_l,
                                               u16* __restrict__ asT_h,
                                               u16* __restrict__ asT_l,
                                               float* __restrict__ d_flat) {
    __shared__ float Asm[2][64 * 36];
    __shared__ u16 Bh[2][64 * 40];
    __shared__ u16 Bl[2][64 * 40];
    __shared__ float rsum[64];
    int t = threadIdx.x, lane = t & 31, w = t >> 5;
    int strip = blockIdx.x, b = blockIdx.y;
    const bool strip_ok = (strip < 31);     // rows strip*64..+63 all < 2000
    int rt = w & 3;                         // row tile (16 rows)
    int ch = w >> 2;                        // column half (2x16 cols)
    float dsum[2] = {0, 0};
    v8f zero = {};
    v8f acc[2] = {zero, zero};
    const size_t adjb = (size_t)b * NN * NN;

    stage_chunk(0, t, strip, b, strip_ok, adj, adjb, sT_h, sT_l,
                &Asm[0][0], &Bh[0][0], &Bl[0][0]);

    for (int c = 0; c < 63; ++c) {
        int buf = c & 1, nbuf = buf ^ 1;
        int pend = 0;
        if (c + 1 < 63)
            pend = stage_chunk(c + 1, t, strip, b, strip_ok, adj, adjb, sT_h, sT_l,
                               &Asm[nbuf][0], &Bh[nbuf][0], &Bl[nbuf][0]);
#if HAVE_ASYNC
        if (pend == 4)      __builtin_amdgcn_s_wait_asynccnt(4);
        else if (pend == 2) __builtin_amdgcn_s_wait_asynccnt(2);
        else                __builtin_amdgcn_s_wait_asynccnt(0);
#else
        (void)pend;
#endif
        __syncthreads();

        // fused degree accumulation (row fixed per (t,q) across chunks)
#pragma unroll
        for (int q = 0; q < 2; ++q) {
            int idx = t + (q << 8);
            float4 v = *(const float4*)&Asm[buf][(idx >> 3) * 36 + ((idx & 7) << 2)];
            dsum[q] += v.x + v.y + v.z + v.w;
        }
        // build split-bf16 A fragment from LDS f32 tile
        {
            int r = (rt << 4) + (lane & 15);
            const float* ar = &Asm[buf][r * 36];
            int koffA = lane & 16;
            Frag ah, al;
#pragma unroll
            for (int i = 0; i < 8; ++i) {
                int k = koffA + (i << 1);
                split2(ar[k], ar[k + 1], ah.u[i], al.u[i]);
            }
#pragma unroll
            for (int ct = 0; ct < 2; ++ct) {
                int n = (ch << 5) + (ct << 4) + (lane & 15);
                int koffB = lane & 16;
                Frag bh, bl;
#pragma unroll
                for (int i = 0; i < 8; ++i) {
                    bh.u[i] = *(const u32*)(&Bh[buf][0] + n * 40 + koffB + (i << 1));
                    bl.u[i] = *(const u32*)(&Bl[buf][0] + n * 40 + koffB + (i << 1));
                }
                acc[ct] = wmma3(ah, al, bh, bl, acc[ct]);
            }
        }
        __syncthreads();
    }

    // store as^T (rows >= NN came from zero-filled adj -> zeros, safe)
    int mhalf = (lane & 16) ? 8 : 0;
#pragma unroll
    for (int ct = 0; ct < 2; ++ct) {
#pragma unroll
        for (int j = 0; j < 8; ++j) {
            int node = (strip << 6) + (rt << 4) + mhalf + j;
            int cl = (ch << 5) + (ct << 4) + (lane & 15);
            float v = acc[ct][j];
            u16 h = f2bf(v);
            size_t o = (size_t)(b * KC + cl) * NP + node;
            asT_h[o] = h;
            asT_l[o] = f2bf(v - bf2f(h));
        }
    }
    // d_flat (degree) via LDS reduce
    if (t < 64) rsum[t] = 0.0f;
    __syncthreads();
#pragma unroll
    for (int q = 0; q < 2; ++q) {
        int idx = t + (q << 8);
        atomicAdd(&rsum[idx >> 3], dsum[q]);
    }
    __syncthreads();
    if (t < 64) d_flat[(size_t)b * NP + (strip << 6) + t] = rsum[t];
}

// ---------------------------------------------------------------- out = s^T x  [B,64,128]
__global__ __launch_bounds__(256) void k_sx(const u16* __restrict__ sT_h,
                                            const u16* __restrict__ sT_l,
                                            const u16* __restrict__ xT_h,
                                            const u16* __restrict__ xT_l,
                                            float* __restrict__ out) {
    int t = threadIdx.x, lane = t & 31, w = t >> 5;
    int b = blockIdx.x;
    int mt = w >> 1, fg = w & 1;
    v8f zero = {};
    v8f acc[4] = {zero, zero, zero, zero};
    size_t arow = (size_t)(b * KC + (mt << 4) + (lane & 15)) * NP;
    for (int kk = 0; kk < KEND; kk += 32) {
        Frag ah, al;
        fragA_load(sT_h + arow + kk, lane, ah);
        fragA_load(sT_l + arow + kk, lane, al);
#pragma unroll
        for (int nt = 0; nt < 4; ++nt) {
            int fr = (fg << 6) + (nt << 4) + (lane & 15);
            size_t brow = (size_t)(b * FF + fr) * NP + kk;
            Frag bh, bl;
            fragB_load(xT_h + brow, lane, bh);
            fragB_load(xT_l + brow, lane, bl);
            acc[nt] = wmma3(ah, al, bh, bl, acc[nt]);
        }
    }
    int mhalf = (lane & 16) ? 8 : 0;
#pragma unroll
    for (int nt = 0; nt < 4; ++nt) {
#pragma unroll
        for (int j = 0; j < 8; ++j) {
            int m = (mt << 4) + mhalf + j;
            int f = (fg << 6) + (nt << 4) + (lane & 15);
            out[(size_t)(b * KC + m) * FF + f] = acc[nt][j];
        }
    }
}

// ---------------------------------------------------------------- out_adj = as^T s, ss = s^T s, losses, normalize
__global__ __launch_bounds__(256) void k_pool(const u16* __restrict__ sT_h,
                                              const u16* __restrict__ sT_l,
                                              const u16* __restrict__ asT_h,
                                              const u16* __restrict__ asT_l,
                                              const float* __restrict__ sden,
                                              const float* __restrict__ d_flat,
                                              float* __restrict__ d_out) {
    __shared__ float oa[64 * 65];
    __shared__ float ssm[64 * 65];
    __shared__ float red[256];
    __shared__ float dvec[64];
    int t = threadIdx.x, lane = t & 31, w = t >> 5;
    int b = blockIdx.x;
    int which = w >> 2, mt = w & 3;  // which==0: out_adj (A=as^T), which==1: ss (A=s^T)
    const u16* Ah = which ? sT_h : asT_h;
    const u16* Al = which ? sT_l : asT_l;
    v8f zero = {};
    v8f acc[4] = {zero, zero, zero, zero};
    size_t arow = (size_t)(b * KC + (mt << 4) + (lane & 15)) * NP;
    for (int kk = 0; kk < KEND; kk += 32) {
        Frag ah, al;
        fragA_load(Ah + arow + kk, lane, ah);
        fragA_load(Al + arow + kk, lane, al);
#pragma unroll
        for (int nt = 0; nt < 4; ++nt) {
            size_t brow = (size_t)(b * KC + (nt << 4) + (lane & 15)) * NP + kk;
            Frag bh, bl;
            fragB_load(sT_h + brow, lane, bh);
            fragB_load(sT_l + brow, lane, bl);
            acc[nt] = wmma3(ah, al, bh, bl, acc[nt]);
        }
    }
    float* dst = which ? ssm : oa;
    int mhalf = (lane & 16) ? 8 : 0;
#pragma unroll
    for (int nt = 0; nt < 4; ++nt)
#pragma unroll
        for (int j = 0; j < 8; ++j)
            dst[((mt << 4) + mhalf + j) * 65 + (nt << 4) + (lane & 15)] = acc[nt][j];
    __syncthreads();

    // mincut loss
    float tr = block_reduce((t < 64) ? oa[t * 66] : 0.0f, red, t);
    float pden = 0.0f;
    for (int n = t; n < NN; n += 256)
        pden += d_flat[(size_t)b * NP + n] * sden[(size_t)b * NP + n];
    float den = block_reduce(pden, red, t);
    if (t == 0) atomicAdd(&d_out[98304], -(tr / den) * 0.125f);

    // ortho loss
    float p1 = 0.0f;
    for (int idx = t; idx < 4096; idx += 256) {
        float v = ssm[(idx >> 6) * 65 + (idx & 63)];
        p1 += v * v;
    }
    float ssq = block_reduce(p1, red, t);
    float invn = 1.0f / sqrtf(ssq);
    float p2 = 0.0f;
    for (int idx = t; idx < 4096; idx += 256) {
        int r = idx >> 6, c = idx & 63;
        float v = ssm[r * 65 + c] * invn - ((r == c) ? 0.125f : 0.0f);
        p2 += v * v;
    }
    float o2 = block_reduce(p2, red, t);
    if (t == 0) atomicAdd(&d_out[98305], sqrtf(o2) * 0.125f);

    // zero diag + symmetric degree normalization
    if (t < 64) {
        float rs = 0.0f;
        for (int l = 0; l < 64; ++l)
            if (l != t) rs += oa[t * 65 + l];
        dvec[t] = sqrtf(rs) + 1e-15f;
    }
    __syncthreads();
    for (int idx = t; idx < 4096; idx += 256) {
        int r = idx >> 6, c = idx & 63;
        float v = (r == c) ? 0.0f : oa[r * 65 + c] / (dvec[r] * dvec[c]);
        d_out[65536 + (size_t)b * 4096 + idx] = v;
    }
}

// ---------------------------------------------------------------- launch
extern "C" void kernel_launch(void* const* d_in, const int* in_sizes, int n_in,
                              void* d_out, int out_size, void* d_ws, size_t ws_size,
                              hipStream_t stream) {
    (void)in_sizes; (void)n_in; (void)out_size; (void)ws_size;
    const float* x   = (const float*)d_in[0];
    const float* adj = (const float*)d_in[1];
    const float* s   = (const float*)d_in[2];
    const unsigned char* mask = (const unsigned char*)d_in[3];
    float* out = (float*)d_out;

    char* w = (char*)d_ws;
    const size_t SZ_ST  = (size_t)BB * KC * NP * sizeof(u16);   // 2 MB
    const size_t SZ_XT  = (size_t)BB * FF * NP * sizeof(u16);   // 4 MB
    u16* sT_h  = (u16*)w;              w += SZ_ST;
    u16* sT_l  = (u16*)w;              w += SZ_ST;
    u16* xT_h  = (u16*)w;              w += SZ_XT;
    u16* xT_l  = (u16*)w;              w += SZ_XT;
    u16* asT_h = (u16*)w;              w += SZ_ST;
    u16* asT_l = (u16*)w;              w += SZ_ST;
    float* s_soft = (float*)w;         w += (size_t)BB * NN * KC * sizeof(float);
    float* sden   = (float*)w;         w += (size_t)BB * NP * sizeof(float);
    float* d_flat = (float*)w;         w += (size_t)BB * NP * sizeof(float);

    k_init<<<1, 32, 0, stream>>>(out);
    k_softmax<<<2000, 256, 0, stream>>>(s, mask, s_soft, sden);
    k_transpose_s<<<dim3(64, 2, 8), 256, 0, stream>>>(s_soft, sT_h, sT_l);
    k_transpose_x<<<dim3(64, 4, 8), 256, 0, stream>>>(x, mask, xT_h, xT_l);
    k_adj_s<<<dim3(32, 8), 256, 0, stream>>>(adj, sT_h, sT_l, asT_h, asT_l, d_flat);
    k_sx<<<8, 256, 0, stream>>>(sT_h, sT_l, xT_h, xT_l, out);
    k_pool<<<8, 256, 0, stream>>>(sT_h, sT_l, asT_h, asT_l, sden, d_flat, out);
}
